// HebrewRootTransformer_90537910600207
// MI455X (gfx1250) — matrix operand order repaired and checked by reading, not compiled, not run on hardware
//
#include <hip/hip_runtime.h>
#include <hip/hip_bf16.h>
#include <cmath>

typedef __attribute__((ext_vector_type(16))) _Float16 v16h;
typedef __attribute__((ext_vector_type(8)))  _Float16 v8h;
typedef __attribute__((ext_vector_type(8)))  float    v8f;

#define NTOK 4096
#define B_   4
#define S_   1024
#define L_   4
#define D_   256
#define H_   8
#define F_   1024
#define HD_  32
#define NLET 22

#define LN1E4 9.2103403719761836f   // ln(10000)

// ---------------- weight convert + transpose: f32 (K x N) -> f16 (N x K) -----
__global__ void k_w2h_t(const float* __restrict__ src, _Float16* __restrict__ dst,
                        int K, int N) {
    int idx = blockIdx.x * blockDim.x + threadIdx.x;
    if (idx >= K * N) return;
    int k = idx / N, n = idx - k * N;
    dst[(size_t)n * K + k] = (_Float16)src[idx];
}

// ---------------- root letter gather: (t) -> rootin16[t][3*256] --------------
__global__ void k_gather_root(const int* __restrict__ roots,
                              const float* __restrict__ emb,
                              _Float16* __restrict__ out) {
    int t = blockIdx.x, d = threadIdx.x;
#pragma unroll
    for (int l = 0; l < 3; ++l) {
        int r = roots[t * 3 + l];
        out[(size_t)t * 768 + l * 256 + d] = (_Float16)emb[(size_t)r * 256 + d];
    }
}

// ---------------- gematria sinusoidal into concat512 cols 256..511 -----------
__global__ void k_gem_embed(const float* __restrict__ gem,
                            _Float16* __restrict__ cat) {
    int t = blockIdx.x, i = threadIdx.x;            // i in 0..127
    float g = gem[t];
    float ang = g * 500.0f * __expf((float)(2 * i) * (-LN1E4 / (float)D_));
    cat[(size_t)t * 512 + 256 + i]       = (_Float16)__sinf(ang);
    cat[(size_t)t * 512 + 256 + 128 + i] = (_Float16)__cosf(ang);
}

// ---------------- LayerNorm (one wave per token) -----------------------------
template<bool ADD_PE>
__global__ void __launch_bounds__(32) k_ln(const float* __restrict__ in,
                    const float* __restrict__ g, const float* __restrict__ bta,
                    float* __restrict__ out32, _Float16* __restrict__ out16,
                    int o16stride, int o16off) {
    int t = blockIdx.x, lane = threadIdx.x;
    float v[8];
    const float* row = in + (size_t)t * D_;
#pragma unroll
    for (int j = 0; j < 8; ++j) v[j] = row[lane * 8 + j];
    if (ADD_PE) {
        int s = t & (S_ - 1);
        float pos = (float)(S_ - 1 - s);            // flipped (RTL) position
#pragma unroll
        for (int j = 0; j < 8; ++j) {
            int d = lane * 8 + j;
            float arg = pos * __expf((float)(d & ~1) * (-LN1E4 / (float)D_));
            v[j] += (d & 1) ? __cosf(arg) : __sinf(arg);
        }
    }
    float s = 0.f;
#pragma unroll
    for (int j = 0; j < 8; ++j) s += v[j];
#pragma unroll
    for (int off = 16; off >= 1; off >>= 1) s += __shfl_xor(s, off, 32);
    float mean = s * (1.0f / D_);
    float q = 0.f;
#pragma unroll
    for (int j = 0; j < 8; ++j) { float d0 = v[j] - mean; q += d0 * d0; }
#pragma unroll
    for (int off = 16; off >= 1; off >>= 1) q += __shfl_xor(q, off, 32);
    float rstd = rsqrtf(q * (1.0f / D_) + 1e-5f);
#pragma unroll
    for (int j = 0; j < 8; ++j) {
        int d = lane * 8 + j;
        float y = (v[j] - mean) * rstd * g[d] + bta[d];
        if (out32) out32[(size_t)t * D_ + d] = y;
        if (out16) out16[(size_t)t * o16stride + o16off + d] = (_Float16)y;
    }
}

__device__ __forceinline__ v16h pack16(v8h lo, v8h hi) {
    v16h a;
#pragma unroll
    for (int i = 0; i < 8; ++i) { a[i] = lo[i]; a[8 + i] = hi[i]; }
    return a;
}

// ---------------- WMMA GEMM, block-tiled with async B staging ----------------
// C = A(MxK,f16) @ W(KxN) (+bias).  Wt stored transposed (N x K) f16.
// Block = 128 threads (4 waves, 2x2), block tile 64(M) x 128(N).
// Each wave: 2 M-subtiles x 4 N-subtiles = 8 WMMA accumulators.
// B tile (128 cols x 32 K halves, 8KB) staged in LDS per K-chunk using
// GLOBAL_LOAD_ASYNC_TO_LDS_B128 (ASYNCcnt) and shared by all 4 waves.
// EPI: 0=f32, 1=f16, 2=V-transpose scatter, 3=residual add f32, 4=gelu->f16
template<int EPI>
__global__ void __launch_bounds__(128) k_gemm(const _Float16* __restrict__ A,
                      const _Float16* __restrict__ Wt,
                      const float* __restrict__ bias,
                      float* __restrict__ outf, _Float16* __restrict__ outh,
                      int K, int N) {
    __shared__ _Float16 lb[128 * 32];                // [col][k], 8 KB
    int tid  = threadIdx.x;
    int lane = tid & 31, w = tid >> 5;
    int half = lane >> 4, l16 = lane & 15;
    int wm = w >> 1, wn = w & 1;
    int bn0 = blockIdx.x * 128;                      // block N base
    int bm0 = blockIdx.y * 64;                       // block M base

    const _Float16* Arow0 = A + (size_t)(bm0 + wm * 32 + l16) * K;
    const _Float16* Arow1 = Arow0 + (size_t)16 * K;

    v8f acc[2][4] = {};
    for (int k0 = 0; k0 < K; k0 += 32) {
        // ---- async-stage B tile: 512 x 16B chunks, 4 per thread ----
#pragma unroll
        for (int i = 0; i < 4; ++i) {
            int c = tid * 4 + i;                     // chunk id 0..511
            int col = c >> 2, part = c & 3;
            const _Float16* gp = Wt + (size_t)(bn0 + col) * K + k0 + part * 8;
            unsigned loff = (unsigned)(size_t)(lb + col * 32 + part * 8);
            asm volatile("global_load_async_to_lds_b128 %0, %1, off"
                         :: "v"(loff), "v"((unsigned long long)(size_t)gp)
                         : "memory");
        }
        // ---- A fragments (direct global) while async copies fly ----
        v16h a0 = *(const v16h*)(Arow0 + k0 + half * 16);
        v16h a1 = *(const v16h*)(Arow1 + k0 + half * 16);
        asm volatile("s_wait_asynccnt 0x0" ::: "memory");
        __syncthreads();                             // B tile visible to block
#pragma unroll
        for (int nt = 0; nt < 4; ++nt) {
            int col = wn * 64 + nt * 16 + l16;       // B-fragment column
            v16h b = *(const v16h*)(lb + col * 32 + half * 16);
            acc[0][nt] = __builtin_amdgcn_wmma_f32_16x16x32_f16(
                             false, a0, false, b, (short)0, acc[0][nt], false, false);
            acc[1][nt] = __builtin_amdgcn_wmma_f32_16x16x32_f16(
                             false, a1, false, b, (short)0, acc[1][nt], false, false);
        }
        __syncthreads();                             // reads done before overwrite
    }
    // ---- epilogue ----
#pragma unroll
    for (int mt = 0; mt < 2; ++mt) {
#pragma unroll
        for (int nt = 0; nt < 4; ++nt) {
            int col = bn0 + wn * 64 + nt * 16 + l16;
            float bv = bias ? bias[col] : 0.0f;
#pragma unroll
            for (int r = 0; r < 8; ++r) {
                int row = bm0 + wm * 32 + mt * 16 + r + 8 * half;
                float v = acc[mt][nt][r] + bv;
                if (EPI == 0) {
                    outf[(size_t)row * N + col] = v;
                } else if (EPI == 1) {
                    outh[(size_t)row * N + col] = (_Float16)v;
                } else if (EPI == 2) {               // V -> [B][H][HD][S]
                    int bb = row >> 10, ss = row & (S_ - 1);
                    int hh = col >> 5,  hd = col & (HD_ - 1);
                    outh[((size_t)((bb * H_ + hh) * HD_ + hd)) * S_ + ss] = (_Float16)v;
                } else if (EPI == 3) {
                    outf[(size_t)row * N + col] += v;
                } else if (EPI == 4) {
                    float ge = 0.5f * v * (1.0f + erff(v * 0.70710678118654752f));
                    outh[(size_t)row * N + col] = (_Float16)ge;
                }
            }
        }
    }
}

// ---------------- attention: 1 wave per (b, h, 16-query tile) ----------------
__global__ void __launch_bounds__(32) k_attn(const _Float16* __restrict__ Q,
                      const _Float16* __restrict__ Km,
                      const _Float16* __restrict__ Vt,   // [B][H][HD][S]
                      _Float16* __restrict__ O,
                      const float* __restrict__ dist_scale,
                      const float* __restrict__ diss_sens) {
    __shared__ _Float16 sc[16 * S_];                 // 32 KB score strip
    __shared__ float rs[16];
    int qt = blockIdx.x, h = blockIdx.y, b = blockIdx.z;
    int lane = threadIdx.x, half = lane >> 4, l16 = lane & 15;
    float pen = -fabsf(dist_scale[h] * (1.0f - 0.5f * diss_sens[h]));
    const float scale = 0.17677669529663687f;        // 1/sqrt(32)

    const _Float16* Arow = Q + ((size_t)(b * S_ + qt * 16 + l16)) * D_ + h * HD_;
    v16h aq = *(const v16h*)(Arow + half * 16);

    // phase 1: scores = Q K^T * scale + distance bias  -> LDS (f16)
    for (int kt = 0; kt < S_ / 16; ++kt) {
        const _Float16* Brow = Km + ((size_t)(b * S_ + kt * 16 + l16)) * D_ + h * HD_;
        v16h bk = *(const v16h*)(Brow + half * 16);
        v8f acc = {};
        acc = __builtin_amdgcn_wmma_f32_16x16x32_f16(
                  false, aq, false, bk, (short)0, acc, false, false);
        int key = kt * 16 + l16;
#pragma unroll
        for (int r = 0; r < 8; ++r) {
            int m = r + 8 * half;
            int qabs = qt * 16 + m;
            float v = acc[r] * scale + pen * fabsf((float)(qabs - key));
            sc[m * S_ + key] = (_Float16)v;
        }
    }
    __syncthreads();

    // phase 2: softmax per row (lane pair l16 / l16+16 split the 1024 columns)
    {
        int m = l16;
        _Float16* rowp = sc + m * S_ + half * 512;
        float mx = -1e30f;
        for (int j = 0; j < 512; ++j) mx = fmaxf(mx, (float)rowp[j]);
        mx = fmaxf(mx, __shfl_xor(mx, 16, 32));
        float sum = 0.f;
        for (int j = 0; j < 512; ++j) {
            float e = __expf((float)rowp[j] - mx);
            sum += e;
            rowp[j] = (_Float16)e;
        }
        sum += __shfl_xor(sum, 16, 32);
        if (half == 0) rs[m] = sum;
    }
    __syncthreads();

    // phase 3: out = P @ V  (N = HD = 32 -> two 16-wide WMMA tiles)
    v8f a0 = {}, a1 = {};
    const _Float16* Vb = Vt + ((size_t)((b * H_ + h) * HD_)) * S_;
    for (int kc = 0; kc < S_ / 32; ++kc) {
        const _Float16* prow = sc + l16 * S_ + kc * 32;
        v16h ap = *(const v16h*)(prow + half * 16);
        v16h b0 = *(const v16h*)(Vb + (size_t)l16 * S_        + kc * 32 + half * 16);
        v16h b1 = *(const v16h*)(Vb + (size_t)(16 + l16) * S_ + kc * 32 + half * 16);
        a0 = __builtin_amdgcn_wmma_f32_16x16x32_f16(
                 false, ap, false, b0, (short)0, a0, false, false);
        a1 = __builtin_amdgcn_wmma_f32_16x16x32_f16(
                 false, ap, false, b1, (short)0, a1, false, false);
    }
#pragma unroll
    for (int r = 0; r < 8; ++r) {
        int m = r + 8 * half;
        float inv = 1.0f / rs[m];
        int q = qt * 16 + m;
        size_t base = ((size_t)(b * S_ + q)) * D_ + h * HD_;
        O[base + l16]      = (_Float16)(a0[r] * inv);
        O[base + 16 + l16] = (_Float16)(a1[r] * inv);
    }
}

// ---------------- output heads (22-wide, VALU) -------------------------------
__global__ void k_heads(const _Float16* __restrict__ X,
                        const float* __restrict__ w1, const float* __restrict__ b1,
                        const float* __restrict__ w2, const float* __restrict__ b2,
                        const float* __restrict__ w3, const float* __restrict__ b3,
                        float* __restrict__ out) {
    int t = blockIdx.x, tid = threadIdx.x;
    if (tid >= 3 * NLET) return;
    int head = tid / NLET, o = tid - head * NLET;
    const float* w  = (head == 0) ? w1 : (head == 1) ? w2 : w3;
    const float* bb = (head == 0) ? b1 : (head == 1) ? b2 : b3;
    const _Float16* x = X + (size_t)t * D_;
    float s = bb[o];
    for (int d = 0; d < D_; ++d) s += (float)x[d] * w[d * NLET + o];
    out[(size_t)head * NTOK * NLET + (size_t)t * NLET + o] = s;
}

// ============================================================================
extern "C" void kernel_launch(void* const* d_in, const int* in_sizes, int n_in,
                              void* d_out, int out_size, void* d_ws, size_t ws_size,
                              hipStream_t stream) {
    const int*   roots        = (const int*)d_in[0];
    const float* gem          = (const float*)d_in[1];
    const float* letter_embed = (const float*)d_in[2];
    const float* root_proj_w  = (const float*)d_in[3];
    const float* root_proj_b  = (const float*)d_in[4];
    const float* root_ln_g    = (const float*)d_in[5];
    const float* root_ln_b    = (const float*)d_in[6];
    const float* input_proj_w = (const float*)d_in[7];
    const float* input_proj_b = (const float*)d_in[8];
    const float* input_ln_g   = (const float*)d_in[9];
    const float* input_ln_b   = (const float*)d_in[10];
    const float* qw = (const float*)d_in[11]; const float* qb = (const float*)d_in[12];
    const float* kw = (const float*)d_in[13]; const float* kb = (const float*)d_in[14];
    const float* vw = (const float*)d_in[15]; const float* vb = (const float*)d_in[16];
    const float* ow = (const float*)d_in[17]; const float* ob = (const float*)d_in[18];
    const float* dist_scale = (const float*)d_in[19];
    const float* diss_sens  = (const float*)d_in[20];
    const float* ff1w = (const float*)d_in[21]; const float* ff1b = (const float*)d_in[22];
    const float* ff2w = (const float*)d_in[23]; const float* ff2b = (const float*)d_in[24];
    const float* ln1g = (const float*)d_in[25]; const float* ln1b = (const float*)d_in[26];
    const float* ln2g = (const float*)d_in[27]; const float* ln2b = (const float*)d_in[28];
    const float* out_ln_g = (const float*)d_in[29]; const float* out_ln_b = (const float*)d_in[30];
    const float* h1w = (const float*)d_in[31]; const float* h1b = (const float*)d_in[32];
    const float* h2w = (const float*)d_in[33]; const float* h2b = (const float*)d_in[34];
    const float* h3w = (const float*)d_in[35]; const float* h3b = (const float*)d_in[36];

    char* ws = (char*)d_ws;
    size_t off = 0;
    auto alloc = [&](size_t bytes) -> void* {
        void* p = ws + off;
        off = (off + bytes + 255) & ~(size_t)255;
        return p;
    };
    _Float16* wt_root  = (_Float16*)alloc((size_t)768 * 256 * 2);
    _Float16* wt_input = (_Float16*)alloc((size_t)512 * 256 * 2);
    _Float16* wt_q     = (_Float16*)alloc((size_t)L_ * D_ * D_ * 2);
    _Float16* wt_k     = (_Float16*)alloc((size_t)L_ * D_ * D_ * 2);
    _Float16* wt_v     = (_Float16*)alloc((size_t)L_ * D_ * D_ * 2);
    _Float16* wt_o     = (_Float16*)alloc((size_t)L_ * D_ * D_ * 2);
    _Float16* wt_ff1   = (_Float16*)alloc((size_t)L_ * D_ * F_ * 2);
    _Float16* wt_ff2   = (_Float16*)alloc((size_t)L_ * F_ * D_ * 2);
    _Float16* rootin16 = (_Float16*)alloc((size_t)NTOK * 768 * 2);
    _Float16* concat16 = (_Float16*)alloc((size_t)NTOK * 512 * 2);
    float*    tmpf     = (float*)   alloc((size_t)NTOK * D_ * 4);
    float*    xf       = (float*)   alloc((size_t)NTOK * D_ * 4);
    _Float16* normed16 = (_Float16*)alloc((size_t)NTOK * D_ * 2);
    _Float16* ff16     = (_Float16*)alloc((size_t)NTOK * F_ * 2);
    _Float16* q16      = (_Float16*)alloc((size_t)NTOK * D_ * 2);
    _Float16* k16      = (_Float16*)alloc((size_t)NTOK * D_ * 2);
    _Float16* vt16     = (_Float16*)alloc((size_t)B_ * H_ * HD_ * S_ * 2);
    _Float16* attn16   = (_Float16*)alloc((size_t)NTOK * D_ * 2);

    auto tconv = [&](const float* src, _Float16* dst, int K, int N) {
        int n = K * N;
        k_w2h_t<<<(n + 255) / 256, 256, 0, stream>>>(src, dst, K, N);
    };
    tconv(root_proj_w,  wt_root,  768, 256);
    tconv(input_proj_w, wt_input, 512, 256);
    for (int l = 0; l < L_; ++l) {
        tconv(qw + (size_t)l * D_ * D_, wt_q + (size_t)l * D_ * D_, D_, D_);
        tconv(kw + (size_t)l * D_ * D_, wt_k + (size_t)l * D_ * D_, D_, D_);
        tconv(vw + (size_t)l * D_ * D_, wt_v + (size_t)l * D_ * D_, D_, D_);
        tconv(ow + (size_t)l * D_ * D_, wt_o + (size_t)l * D_ * D_, D_, D_);
        tconv(ff1w + (size_t)l * D_ * F_, wt_ff1 + (size_t)l * D_ * F_, D_, F_);
        tconv(ff2w + (size_t)l * F_ * D_, wt_ff2 + (size_t)l * F_ * D_, F_, D_);
    }

    // ----- embedding path -----
    k_gather_root<<<NTOK, 256, 0, stream>>>(roots, letter_embed, rootin16);
    k_gem_embed<<<NTOK, 128, 0, stream>>>(gem, concat16);

    dim3 gD(D_ / 128, NTOK / 64);                // (2, 64), 128 thr blocks
    k_gemm<0><<<gD, 128, 0, stream>>>(rootin16, wt_root, root_proj_b,
                                      tmpf, nullptr, 768, D_);
    k_ln<false><<<NTOK, 32, 0, stream>>>(tmpf, root_ln_g, root_ln_b,
                                         nullptr, concat16, 512, 0);
    k_gemm<0><<<gD, 128, 0, stream>>>(concat16, wt_input, input_proj_b,
                                      tmpf, nullptr, 512, D_);
    k_ln<true><<<NTOK, 32, 0, stream>>>(tmpf, input_ln_g, input_ln_b,
                                        xf, nullptr, 0, 0);

    // ----- transformer layers -----
    for (int l = 0; l < L_; ++l) {
        k_ln<false><<<NTOK, 32, 0, stream>>>(xf, ln1g + l * D_, ln1b + l * D_,
                                             nullptr, normed16, D_, 0);
        k_gemm<1><<<gD, 128, 0, stream>>>(normed16, wt_q + (size_t)l * D_ * D_,
                                          qb + l * D_, nullptr, q16, D_, D_);
        k_gemm<1><<<gD, 128, 0, stream>>>(normed16, wt_k + (size_t)l * D_ * D_,
                                          kb + l * D_, nullptr, k16, D_, D_);
        k_gemm<2><<<gD, 128, 0, stream>>>(normed16, wt_v + (size_t)l * D_ * D_,
                                          vb + l * D_, nullptr, vt16, D_, D_);
        dim3 ga(S_ / 16, H_, B_);
        k_attn<<<ga, 32, 0, stream>>>(q16, k16, vt16, attn16,
                                      dist_scale + l * H_, diss_sens + l * H_);
        k_gemm<3><<<gD, 128, 0, stream>>>(attn16, wt_o + (size_t)l * D_ * D_,
                                          ob + l * D_, xf, nullptr, D_, D_);
        k_ln<false><<<NTOK, 32, 0, stream>>>(xf, ln2g + l * D_, ln2b + l * D_,
                                             nullptr, normed16, D_, 0);
        dim3 gF(F_ / 128, NTOK / 64);            // (8, 64)
        k_gemm<4><<<gF, 128, 0, stream>>>(normed16, wt_ff1 + (size_t)l * D_ * F_,
                                          ff1b + l * F_, nullptr, ff16, D_, F_);
        k_gemm<3><<<gD, 128, 0, stream>>>(ff16, wt_ff2 + (size_t)l * F_ * D_,
                                          ff2b + l * D_, xf, nullptr, F_, D_);
    }

    // ----- output -----
    k_ln<false><<<NTOK, 32, 0, stream>>>(xf, out_ln_g, out_ln_b,
                                         nullptr, normed16, D_, 0);
    k_heads<<<NTOK, 96, 0, stream>>>(normed16, h1w, h1b, h2w, h2b, h3w, h3b,
                                     (float*)d_out);
    (void)in_sizes; (void)n_in; (void)out_size; (void)ws_size;
}
